// SSPPI_74380243632833
// MI455X (gfx1250) — matrix operand order repaired and checked by reading, not compiled, not run on hardware
//
#include <hip/hip_runtime.h>
#include <hip/hip_bf16.h>
#include <math.h>

typedef __attribute__((ext_vector_type(16))) _Float16 v16h;
typedef __attribute__((ext_vector_type(8)))  _Float16 v8h;
typedef __attribute__((ext_vector_type(8)))  float    v8f;
typedef __attribute__((ext_vector_type(4)))  unsigned int v4u;
typedef __attribute__((ext_vector_type(8)))  int v8i;
typedef __attribute__((ext_vector_type(4)))  int v4i;

#define SEQ   45
#define SP    48          // padded sequence (3 M-tiles of 16)
#define DM    64
#define NH    8
#define FFD   256
#define NTHR  256
#define NWAVE 8
#define NB    2           // batch elements per workgroup

// ws (f16) layout: WqT WkT WvT WoT (4096 each) | W1T [256][64] | W2T [64][256]
#define OFF_WO  (3*4096)
#define OFF_W1  (4*4096)
#define OFF_W2  (OFF_W1 + 16384)
#define W_TOTAL (OFF_W2 + 16384)   // 49152 halfs = 96 KB

struct __align__(16) Smem {
  _Float16 W[W_TOTAL];              // 96 KB
  _Float16 A16[NB][SP*DM];          // f16 A operand: h -> attn-out -> h2
  _Float16 Q16[NB][SP*DM];
  _Float16 K16[NB][SP*DM];
  _Float16 Vt16[NB][DM*SP];         // [d][token]
  _Float16 Pw[NWAVE][SP*SP];        // wave-private softmax probabilities
  _Float16 F16[NB][SP*FFD];         // FFN hidden (post-ReLU)
  float    Hf[NB][SP*DM];           // f32 running buffer
  float    bq[DM], bk[DM], bv[DM], bo[DM], b2v[DM];
  float    b1v[FFD];
  float    g1[DM], be1[DM], g2[DM], be2[DM];
};                                   // ~255 KB total

__device__ inline v8h v8h_zero() {
  v8h z;
  #pragma unroll
  for (int i = 0; i < 8; ++i) z[i] = (_Float16)0;
  return z;
}
__device__ inline v8f v8f_zero() {
  v8f z;
  #pragma unroll
  for (int i = 0; i < 8; ++i) z[i] = 0.f;
  return z;
}
__device__ inline v16h mk16(v8h a, v8h b) {
  v16h r;
  #pragma unroll
  for (int i = 0; i < 8; ++i) { r[i] = a[i]; r[8+i] = b[i]; }
  return r;
}

// A fragment (16x32 f16, MxK). ISA layout: lane L row M=L&15, group g=L>>4;
// halves 0..7 -> K = k0 + g*8 + j ; halves 8..15 -> K = k0 + 16 + g*8 + (j-8).
__device__ inline v16h ldsA(const _Float16* base, int ld, int m0, int k0,
                            int kLimit, int lane) {
  int r = m0 + (lane & 15);
  int g = lane >> 4;
  int ka = k0 + g * 8;
  int kb = k0 + 16 + g * 8;
  const _Float16* row = base + r * ld;
  v8h x0 = (ka < kLimit) ? *(const v8h*)(row + ka) : v8h_zero();
  v8h x1 = (kb < kLimit) ? *(const v8h*)(row + kb) : v8h_zero();
  return mk16(x0, x1);
}

// B fragment (32x16 f16, KxN) from N-major storage Bt[N][K].
// ISA layout: lane L col N=L&15, group g=L>>4; halves 0..7 -> K = k0 + g*8 + j ; halves 8..15 -> K = k0 + 16 + g*8 + (j-8).
__device__ inline v16h ldsB(const _Float16* baseT, int ld, int n0, int k0,
                            int kLimit, int nLimit, int lane) {
  int n = n0 + (lane & 15);
  int g = lane >> 4;
  int ka = k0 + g * 8;
  v8h x0 = v8h_zero(), x1 = v8h_zero();
  if (n < nLimit) {
    const _Float16* row = baseT + n * ld;
    if (ka     < kLimit) x0 = *(const v8h*)(row + ka);
    if (ka + 16 < kLimit) x1 = *(const v8h*)(row + ka + 16);
  }
  return mk16(x0, x1);
}

__device__ inline v8f wmma_f16(v16h a, v16h b, v8f c) {
  return __builtin_amdgcn_wmma_f32_16x16x32_f16(false, a, false, b,
                                                (short)0, c, false, false);
}

// ---- weight prep: fp32 -> f16, transposed to [N][K] -------------------------
__global__ void enc_prep_weights(const float* __restrict__ Wq, const float* __restrict__ Wk,
                                 const float* __restrict__ Wv, const float* __restrict__ Wo,
                                 const float* __restrict__ W1, const float* __restrict__ W2,
                                 _Float16* __restrict__ ws) {
  int i = blockIdx.x * NTHR + threadIdx.x;
  if (i >= W_TOTAL) return;
  if (i < OFF_W1) {                       // Wq/Wk/Wv/Wo: [64][64] -> T
    int which = i >> 12, t = i & 4095, n = t >> 6, k = t & 63;
    const float* W = (which == 0) ? Wq : (which == 1) ? Wk : (which == 2) ? Wv : Wo;
    ws[i] = (_Float16)W[k * DM + n];
  } else if (i < OFF_W2) {                // W1 [64][256] -> W1T [256][64]
    int t = i - OFF_W1, n = t >> 6, k = t & 63;
    ws[i] = (_Float16)W1[k * FFD + n];
  } else {                                // W2 [256][64] -> W2T [64][256]
    int t = i - OFF_W2, n = t >> 8, k = t & 255;
    ws[i] = (_Float16)W2[k * DM + n];
  }
}

// ---- fully fused encoder layer: NB batch elements per workgroup -------------
__global__ void __launch_bounds__(NTHR)
enc_layer_fused(const float* __restrict__ x,
                const float* __restrict__ bq, const float* __restrict__ bk,
                const float* __restrict__ bv, const float* __restrict__ bo,
                const float* __restrict__ ln1g, const float* __restrict__ ln1b,
                const float* __restrict__ b1,  const float* __restrict__ b2,
                const float* __restrict__ ln2g, const float* __restrict__ ln2b,
                const _Float16* __restrict__ ws, float* __restrict__ out, int nbatch) {
  __shared__ Smem sm;
  const int tid  = threadIdx.x;
  const int lane = tid & 31;
  const int wave = tid >> 5;
  const int b    = blockIdx.x;
  const float rs8 = 0.35355339059327373f;   // 1/sqrt(head_dim)

  // ---- stage 0: weights via TDM (async DMA) overlapped with PE compute ----
#if __has_builtin(__builtin_amdgcn_tensor_load_to_lds)
  const bool use_tdm = true;
  if (wave == 0) {
    // D# per CDNA5 ISA ch.8: one 2D tile, 1 row of 12288 x 8-byte elements.
    unsigned ldsoff = (unsigned)(size_t)(void*)&sm.W[0];
    unsigned long long ga = (unsigned long long)(size_t)ws;
    v4u g0 = { 1u,                      // count=1, user descriptor
               ldsoff,                  // lds_addr (bits 63:32)
               (unsigned)ga,            // global_addr lo
               (unsigned)(ga >> 32) | 0x80000000u };  // ga hi | type=2 (bits 127:126)
    v8i g1 = { 0x30000,                 // wg_mask=0, data_size=3 (8 bytes)
               0x30000000,              // tensor_dim0[15:0]=12288 @ bits 63:48
               0x00010000,              // tensor_dim0 hi=0, tensor_dim1[15:0]=1
               0x30000000,              // tensor_dim1 hi=0, tile_dim0=12288
               1,                       // tile_dim1=1, tile_dim2=0
               12288,                   // tensor_dim0_stride lo32
               0x30000000,              // stride0 hi=0, tensor_dim1_stride[15:0]=12288
               0 };
    v4i gz = { 0, 0, 0, 0 };            // groups 2/3: dims 3/4 unused
#if defined(__clang_major__) && __clang_major__ >= 23
    v8i gz8 = { 0, 0, 0, 0, 0, 0, 0, 0 };
    __builtin_amdgcn_tensor_load_to_lds(g0, g1, gz, gz, gz8, 0);
#else
    __builtin_amdgcn_tensor_load_to_lds(g0, g1, gz, gz, 0);
#endif
  }
#else
  const bool use_tdm = false;
  {
    const uint4* src = (const uint4*)ws;
    uint4* dst = (uint4*)sm.W;
    #pragma unroll 4
    for (int i = tid; i < W_TOTAL / 8; i += NTHR) dst[i] = src[i];
  }
#endif
  {
    if (tid < DM) {
      sm.bq[tid] = bq[tid];  sm.bk[tid] = bk[tid];
      sm.bv[tid] = bv[tid];  sm.bo[tid] = bo[tid];
      sm.b2v[tid] = b2[tid];
      sm.g1[tid] = ln1g[tid]; sm.be1[tid] = ln1b[tid];
      sm.g2[tid] = ln2g[tid]; sm.be2[tid] = ln2b[tid];
    }
    sm.b1v[tid] = b1[tid];   // tid covers 0..255 == FFD
    const float ln1e4 = 9.210340371976184f;  // log(10000)
    for (int i = tid; i < NB * SP * DM; i += NTHR) {
      int bi = i / (SP * DM), j = i % (SP * DM);
      int s = j >> 6, d = j & 63;
      int gb = b * NB + bi;
      float v = 0.f;
      if (s < SEQ && gb < nbatch) {
        float xv  = x[(size_t)gb * SEQ * DM + s * DM + d];
        float div = __expf(-ln1e4 * (float)(d & ~1) * (1.f / DM));
        float ang = (float)s * div;
        float pe  = (d & 1) ? __cosf(ang) : __sinf(ang);
        v = xv + pe;
      }
      sm.Hf[bi][j]  = v;
      sm.A16[bi][j] = (_Float16)v;
    }
    // speculative prefetch of next block's activations (global_prefetch_b8)
    if (b + 1 < gridDim.x) {
      const float* nx = x + (size_t)(b + 1) * NB * SEQ * DM;
      for (int i = tid; i < (NB * SEQ * DM) / 64; i += NTHR)
        __builtin_prefetch(nx + i * 64, 0, 1);
    }
  }
  if (use_tdm && wave == 0) {
#if __has_builtin(__builtin_amdgcn_s_wait_tensorcnt)
    __builtin_amdgcn_s_wait_tensorcnt(0);
#else
    asm volatile("s_wait_tensorcnt 0x0" ::: "memory");
#endif
  }
  __syncthreads();   // publishes TDM-filled sm.W to all waves

  // ---- stage 1: Q/K/V projections (72 tiles, 9 per wave) ----
  for (int t = wave; t < 3 * NB * 12; t += NWAVE) {
    int which = t / (NB * 12);           // 0=Q, 1=K, 2=V
    int rem   = t % (NB * 12);
    int bi    = rem / 12;
    int tt    = rem % 12;
    int mt = tt >> 2, nt = tt & 3;
    const _Float16* Wt = sm.W + which * 4096;
    v8f acc = v8f_zero();
    #pragma unroll
    for (int ks = 0; ks < 2; ++ks) {
      v16h a  = ldsA(sm.A16[bi], DM, mt * 16, ks * 32, DM, lane);
      v16h bf = ldsB(Wt, DM, nt * 16, ks * 32, DM, DM, lane);
      acc = wmma_f16(a, bf, acc);
    }
    int n = lane & 15, g = lane >> 4;
    int col = nt * 16 + n, row0 = mt * 16 + g * 8;
    if (which == 0) {
      float bb = sm.bq[col];
      #pragma unroll
      for (int r = 0; r < 8; ++r)
        sm.Q16[bi][(row0 + r) * DM + col] = (_Float16)((acc[r] + bb) * rs8);
    } else if (which == 1) {
      float bb = sm.bk[col];
      #pragma unroll
      for (int r = 0; r < 8; ++r)
        sm.K16[bi][(row0 + r) * DM + col] = (_Float16)(acc[r] + bb);
    } else {
      float bb = sm.bv[col];
      #pragma unroll
      for (int r = 0; r < 8; ++r)
        sm.Vt16[bi][col * SP + row0 + r] = (_Float16)(acc[r] + bb);
    }
  }
  __syncthreads();

  // ---- stage 2: fused attention per wave (wave == head; NB batches serial),
  //      P buffer is wave-private -> no workgroup barrier needed inside ----
  {
    const int h = wave;
    _Float16* P = sm.Pw[wave];
    for (int bi = 0; bi < NB; ++bi) {
      // scores + softmax -> P
      for (int mt = 0; mt < 3; ++mt) {
        v8f sc[3];
        v16h a = ldsA(sm.Q16[bi] + h * 8, DM, mt * 16, 0, 8, lane);
        #pragma unroll
        for (int nt = 0; nt < 3; ++nt) {
          v16h bf = ldsB(sm.K16[bi] + h * 8, DM, nt * 16, 0, 8, SP, lane);
          sc[nt] = wmma_f16(a, bf, v8f_zero());
        }
        int cb = lane & 15, g = lane >> 4;
        bool v1ok = (16 + cb) < SEQ, v2ok = (32 + cb) < SEQ;
        #pragma unroll
        for (int r = 0; r < 8; ++r) {
          float s0 = sc[0][r];
          float s1 = v1ok ? sc[1][r] : -1e30f;
          float s2 = v2ok ? sc[2][r] : -1e30f;
          float mx = fmaxf(s0, fmaxf(s1, s2));
          #pragma unroll
          for (int off = 8; off > 0; off >>= 1) mx = fmaxf(mx, __shfl_xor(mx, off, 16));
          float e0 = __expf(s0 - mx);
          float e1 = v1ok ? __expf(s1 - mx) : 0.f;
          float e2 = v2ok ? __expf(s2 - mx) : 0.f;
          float sum = e0 + e1 + e2;
          #pragma unroll
          for (int off = 8; off > 0; off >>= 1) sum += __shfl_xor(sum, off, 16);
          float inv = 1.f / sum;
          _Float16* prow = P + (mt * 16 + g * 8 + r) * SP;
          prow[cb]      = (_Float16)(e0 * inv);
          prow[16 + cb] = (_Float16)(e1 * inv);
          prow[32 + cb] = (_Float16)(e2 * inv);
        }
      }
      // attn @ V -> A16[bi] (same wave reads its own P; LDS is in-order per wave)
      for (int mt = 0; mt < 3; ++mt) {
        v8f acc = v8f_zero();
        #pragma unroll
        for (int ks = 0; ks < 2; ++ks) {
          v16h a  = ldsA(P, SP, mt * 16, ks * 32, SP, lane);
          v16h bf = ldsB(sm.Vt16[bi] + h * 8 * SP, SP, 0, ks * 32, SP, 8, lane);
          acc = wmma_f16(a, bf, acc);
        }
        int n = lane & 15, g = lane >> 4;
        if (n < 8) {
          #pragma unroll
          for (int r = 0; r < 8; ++r)
            sm.A16[bi][(mt * 16 + g * 8 + r) * DM + h * 8 + n] = (_Float16)acc[r];
        }
      }
    }
  }
  __syncthreads();

  // ---- stage 3: output projection + residual into Hf (24 tiles, 3/wave) ----
  for (int t = wave; t < NB * 12; t += NWAVE) {
    int bi = t / 12, tt = t % 12, mt = tt >> 2, nt = tt & 3;
    v8f acc = v8f_zero();
    #pragma unroll
    for (int ks = 0; ks < 2; ++ks) {
      v16h a  = ldsA(sm.A16[bi], DM, mt * 16, ks * 32, DM, lane);
      v16h bf = ldsB(sm.W + OFF_WO, DM, nt * 16, ks * 32, DM, DM, lane);
      acc = wmma_f16(a, bf, acc);
    }
    int n = lane & 15, g = lane >> 4;
    int col = nt * 16 + n, row0 = mt * 16 + g * 8;
    float bb = sm.bo[col];
    #pragma unroll
    for (int r = 0; r < 8; ++r)
      sm.Hf[bi][(row0 + r) * DM + col] += acc[r] + bb;
  }
  __syncthreads();

  // ---- stage 4: LayerNorm 1 (Hf in place, f16 copy into A16) ----
  if (tid < NB * SEQ) {
    int bi = tid / SEQ, tk = tid % SEQ;
    float* row = sm.Hf[bi] + tk * DM;
    float s = 0.f, s2 = 0.f;
    #pragma unroll 8
    for (int d = 0; d < DM; ++d) { float v = row[d]; s += v; s2 += v * v; }
    float m   = s  * (1.f / DM);
    float var = s2 * (1.f / DM) - m * m;
    float inv = rsqrtf(var + 1e-5f);
    #pragma unroll 8
    for (int d = 0; d < DM; ++d) {
      float y = (row[d] - m) * inv * sm.g1[d] + sm.be1[d];
      row[d] = y;
      sm.A16[bi][tk * DM + d] = (_Float16)y;
    }
  }
  __syncthreads();

  // ---- stage 5: FFN1 + ReLU -> F16 (A fragments hoisted per (bi,mt)) ----
  for (int bi = 0; bi < NB; ++bi) {
    for (int mt = 0; mt < 3; ++mt) {
      v16h a0 = ldsA(sm.A16[bi], DM, mt * 16, 0,  DM, lane);
      v16h a1 = ldsA(sm.A16[bi], DM, mt * 16, 32, DM, lane);
      for (int nt = wave; nt < 16; nt += NWAVE) {
        v8f acc = v8f_zero();
        acc = wmma_f16(a0, ldsB(sm.W + OFF_W1, DM, nt * 16, 0,  DM, FFD, lane), acc);
        acc = wmma_f16(a1, ldsB(sm.W + OFF_W1, DM, nt * 16, 32, DM, FFD, lane), acc);
        int n = lane & 15, g = lane >> 4;
        int col = nt * 16 + n, row0 = mt * 16 + g * 8;
        float bb = sm.b1v[col];
        #pragma unroll
        for (int r = 0; r < 8; ++r)
          sm.F16[bi][(row0 + r) * FFD + col] = (_Float16)fmaxf(acc[r] + bb, 0.f);
      }
    }
  }
  __syncthreads();

  // ---- stage 6: FFN2 + residual into Hf (24 tiles, 3/wave) ----
  for (int t = wave; t < NB * 12; t += NWAVE) {
    int bi = t / 12, tt = t % 12, mt = tt >> 2, nt = tt & 3;
    v8f acc = v8f_zero();
    #pragma unroll
    for (int ks = 0; ks < 8; ++ks) {
      v16h a  = ldsA(sm.F16[bi], FFD, mt * 16, ks * 32, FFD, lane);
      v16h bf = ldsB(sm.W + OFF_W2, FFD, nt * 16, ks * 32, FFD, DM, lane);
      acc = wmma_f16(a, bf, acc);
    }
    int n = lane & 15, g = lane >> 4;
    int col = nt * 16 + n, row0 = mt * 16 + g * 8;
    float bb = sm.b2v[col];
    #pragma unroll
    for (int r = 0; r < 8; ++r)
      sm.Hf[bi][(row0 + r) * DM + col] += acc[r] + bb;
  }
  __syncthreads();

  // ---- stage 7: LayerNorm 2 ----
  if (tid < NB * SEQ) {
    int bi = tid / SEQ, tk = tid % SEQ;
    float* row = sm.Hf[bi] + tk * DM;
    float s = 0.f, s2 = 0.f;
    #pragma unroll 8
    for (int d = 0; d < DM; ++d) { float v = row[d]; s += v; s2 += v * v; }
    float m   = s  * (1.f / DM);
    float var = s2 * (1.f / DM) - m * m;
    float inv = rsqrtf(var + 1e-5f);
    #pragma unroll 8
    for (int d = 0; d < DM; ++d)
      row[d] = (row[d] - m) * inv * sm.g2[d] + sm.be2[d];
  }
  __syncthreads();

  // ---- stage 8: coalesced writeout (first SEQ*DM floats of each Hf[bi]) ----
  for (int i = tid; i < NB * SEQ * DM; i += NTHR) {
    int bi = i / (SEQ * DM), r = i % (SEQ * DM);
    int gb = b * NB + bi;
    if (gb < nbatch) out[(size_t)gb * SEQ * DM + r] = sm.Hf[bi][r];
  }
}

extern "C" void kernel_launch(void* const* d_in, const int* in_sizes, int n_in,
                              void* d_out, int out_size, void* d_ws, size_t ws_size,
                              hipStream_t stream) {
  (void)n_in; (void)out_size; (void)ws_size;
  const float* x    = (const float*)d_in[0];
  const float* Wq   = (const float*)d_in[1];
  const float* bq   = (const float*)d_in[2];
  const float* Wk   = (const float*)d_in[3];
  const float* bk   = (const float*)d_in[4];
  const float* Wv   = (const float*)d_in[5];
  const float* bv   = (const float*)d_in[6];
  const float* Wo   = (const float*)d_in[7];
  const float* bo   = (const float*)d_in[8];
  const float* ln1g = (const float*)d_in[9];
  const float* ln1b = (const float*)d_in[10];
  const float* W1   = (const float*)d_in[11];
  const float* b1   = (const float*)d_in[12];
  const float* W2   = (const float*)d_in[13];
  const float* b2   = (const float*)d_in[14];
  const float* ln2g = (const float*)d_in[15];
  const float* ln2b = (const float*)d_in[16];
  _Float16* ws = (_Float16*)d_ws;
  float* out = (float*)d_out;

  int nb = in_sizes[0] / (SEQ * DM);   // 8192 batch elements
  int grid = (nb + NB - 1) / NB;       // 4096 workgroups

  enc_prep_weights<<<(W_TOTAL + NTHR - 1) / NTHR, NTHR, 0, stream>>>(
      Wq, Wk, Wv, Wo, W1, W2, ws);
  enc_layer_fused<<<grid, NTHR, 0, stream>>>(
      x, bq, bk, bv, bo, ln1g, ln1b, b1, b2, ln2g, ln2b, ws, out, nb);
}
